// TransformerTimeSeries_31215822307335
// MI455X (gfx1250) — compile-verified, matrix operand order checked by reading, not run on hardware
//
#include <hip/hip_runtime.h>
#include <hip/hip_bf16.h>
#include <stdint.h>

// ---- problem constants ------------------------------------------------------
#define NB   16
#define NE   25
#define NT   300
#define NCH  3
#define DM   64
#define NH   8
#define HDM  8
#define FFD  256
#define NLAY 4
#define NCLS 60
#define NSEQ 301                   // T+1
#define NROW (NB*NE)               // 400
#define MTOT (NROW*NSEQ)           // 120400
#define EPSLN 1e-5f

typedef __attribute__((ext_vector_type(16))) _Float16 v16h;
typedef __attribute__((ext_vector_type(8)))  float    v8f;

union Frag { v16h v; uint32_t u[8]; _Float16 h[16]; };

// dword j of a 16-bit A/B fragment holds K-pair starting at this K offset
__device__ __forceinline__ int kb_of(int j, int hi) {
    return ((j < 4) ? (j << 1) : (16 + ((j - 4) << 1))) + (hi << 3);
}

__device__ __forceinline__ float wred_sum(float v) {
#pragma unroll
    for (int m = 1; m < 32; m <<= 1) v += __shfl_xor(v, m, 32);
    return v;
}

__device__ __forceinline__ float geluf(float x) {
    return 0.5f * x * (1.0f + erff(x * 0.70710678118654752f));
}

__device__ __forceinline__ float pe_val(int p, int d) {
    int i = d >> 1;
    float freq = __expf(-(float)(2 * i) * (9.210340371976184f / 64.0f));
    float a = (float)p * freq;
    return (d & 1) ? __cosf(a) : __sinf(a);
}

// ---- f32 -> f16 weight conversion ------------------------------------------
__global__ void cvt_f16_kernel(const float* __restrict__ src,
                               _Float16* __restrict__ dst, int n) {
    int i = blockIdx.x * 256 + threadIdx.x;
    if (i < n) dst[i] = (_Float16)src[i];
}

// ---- embed + LN + cls + positional encoding --------------------------------
__global__ __launch_bounds__(256) void embed_ln_pos_kernel(
    const float* __restrict__ x, const float* __restrict__ ew,
    const float* __restrict__ eb, const float* __restrict__ elg,
    const float* __restrict__ elb, const float* __restrict__ cls,
    float* __restrict__ h32) {
    int lane = threadIdx.x & 31, wave = threadIdx.x >> 5;
    int row = blockIdx.x * 8 + wave;
    if (row >= MTOT) return;
    int n = row / NSEQ, s = row - n * NSEQ;
    int d0 = lane, d1 = lane + 32;
    float v0, v1;
    if (s == 0) {
        v0 = cls[d0]; v1 = cls[d1];
    } else {
        int t = s - 1, b = n / NE, e = n - b * NE;
        float x0 = x[((b * NCH + 0) * NT + t) * NE + e];
        float x1 = x[((b * NCH + 1) * NT + t) * NE + e];
        float x2 = x[((b * NCH + 2) * NT + t) * NE + e];
        v0 = eb[d0] + x0 * ew[d0 * NCH] + x1 * ew[d0 * NCH + 1] + x2 * ew[d0 * NCH + 2];
        v1 = eb[d1] + x0 * ew[d1 * NCH] + x1 * ew[d1 * NCH + 1] + x2 * ew[d1 * NCH + 2];
        float mu = wred_sum(v0 + v1) * (1.0f / 64.0f);
        float a0 = v0 - mu, a1 = v1 - mu;
        float var = wred_sum(a0 * a0 + a1 * a1) * (1.0f / 64.0f);
        float rs = rsqrtf(var + EPSLN);
        v0 = a0 * rs * elg[d0] + elb[d0];
        v1 = a1 * rs * elg[d1] + elb[d1];
    }
    v0 += pe_val(s, d0);
    v1 += pe_val(s, d1);
    h32[(size_t)row * DM + d0] = v0;
    h32[(size_t)row * DM + d1] = v1;
}

// ---- layernorm over D=64, f32 in -> f16 out --------------------------------
__global__ __launch_bounds__(256) void ln_kernel(
    const float* __restrict__ hin, const float* __restrict__ g,
    const float* __restrict__ b, _Float16* __restrict__ y16) {
    int lane = threadIdx.x & 31, wave = threadIdx.x >> 5;
    int row = blockIdx.x * 8 + wave;
    if (row >= MTOT) return;
    float v0 = hin[(size_t)row * DM + lane];
    float v1 = hin[(size_t)row * DM + lane + 32];
    float mu = wred_sum(v0 + v1) * (1.0f / 64.0f);
    float a0 = v0 - mu, a1 = v1 - mu;
    float var = wred_sum(a0 * a0 + a1 * a1) * (1.0f / 64.0f);
    float rs = rsqrtf(var + EPSLN);
    y16[(size_t)row * DM + lane]      = (_Float16)(a0 * rs * g[lane] + b[lane]);
    y16[(size_t)row * DM + lane + 32] = (_Float16)(a1 * rs * g[lane + 32] + b[lane + 32]);
}

// ---- WMMA GEMM: C[M,Nout] = A[M,K] * W[Nout,K]^T (+bias, epilogue) ---------
// Block: 128 threads / 4 waves; tile 128 rows x 64 cols.
// Each wave owns 32 rows (two 16-row A frags) so every B fragment feeds 2 WMMAs.
// EPI 0: bias -> f16 out.  EPI 1: bias + residual -> f32 out.  EPI 2: bias+GELU -> f16 out.
template <int K, int EPI>
__global__ __launch_bounds__(128) void gemm_wmma_kernel(
    const _Float16* __restrict__ A, const _Float16* __restrict__ W,
    const float* __restrict__ bias, const float* __restrict__ resid,
    _Float16* __restrict__ out16, float* __restrict__ out32,
    int M, int Nout) {
    int lane = threadIdx.x & 31;
    int wave = threadIdx.x >> 5;
    int hi   = lane >> 4;
    int ln15 = lane & 15;
    int m0 = blockIdx.x * 128 + wave * 32;
    int n0 = blockIdx.y * 64;

    v8f acc[2][4] = {};
    int arow0 = m0 + ln15;
    int arow1 = m0 + 16 + ln15;
    bool ok0 = arow0 < M, ok1 = arow1 < M;
    const _Float16* Ar0 = A + (size_t)(ok0 ? arow0 : 0) * K;
    const _Float16* Ar1 = A + (size_t)(ok1 ? arow1 : 0) * K;

    for (int kk = 0; kk < K; kk += 32) {
        if (K > 64 && kk + 32 < K) {
            __builtin_prefetch(Ar0 + kk + 32, 0, 3);   // global_prefetch_b8
            __builtin_prefetch(Ar1 + kk + 32, 0, 3);
        }
        Frag a0, a1;
#pragma unroll
        for (int j = 0; j < 8; ++j) {
            int kb = kb_of(j, hi) + kk;
            a0.u[j] = ok0 ? *(const uint32_t*)(Ar0 + kb) : 0u;
            a1.u[j] = ok1 ? *(const uint32_t*)(Ar1 + kb) : 0u;
        }
#pragma unroll
        for (int f = 0; f < 4; ++f) {
            int col = n0 + f * 16 + ln15;
            const _Float16* Wrow = W + (size_t)col * K;
            Frag bf;
#pragma unroll
            for (int j = 0; j < 8; ++j) {
                int kb = kb_of(j, hi) + kk;
                bf.u[j] = *(const uint32_t*)(Wrow + kb);
            }
            acc[0][f] = __builtin_amdgcn_wmma_f32_16x16x32_f16(
                false, a0.v, false, bf.v, (short)0, acc[0][f], false, false);
            acc[1][f] = __builtin_amdgcn_wmma_f32_16x16x32_f16(
                false, a1.v, false, bf.v, (short)0, acc[1][f], false, false);
        }
    }
    // epilogue: C layout — vgpr r, lane L: row = r + 8*(L>>4), col = L&15
#pragma unroll
    for (int g2 = 0; g2 < 2; ++g2) {
#pragma unroll
        for (int f = 0; f < 4; ++f) {
            int col = n0 + f * 16 + ln15;
            float bv = bias[col];
#pragma unroll
            for (int r = 0; r < 8; ++r) {
                int row = m0 + g2 * 16 + r + 8 * hi;
                if (row < M) {
                    float v = acc[g2][f][r] + bv;
                    if (EPI == 1) {
                        v += resid[(size_t)row * Nout + col];
                        out32[(size_t)row * Nout + col] = v;
                    } else if (EPI == 2) {
                        out16[(size_t)row * Nout + col] = (_Float16)geluf(v);
                    } else {
                        out16[(size_t)row * Nout + col] = (_Float16)v;
                    }
                }
            }
        }
    }
}

// ---- fused multi-head attention: one block (4 waves) per (n, h) ------------
__global__ __launch_bounds__(128) void attn_kernel(
    const _Float16* __restrict__ qkv,   // [N*S, 192]  (q|k|v)
    _Float16* __restrict__ o16) {       // [N*S, 64]
    __shared__ __align__(16) _Float16 Kt[320 * 8];     // [key][d]
    __shared__ __align__(16) _Float16 Vt[8 * 320];     // [d][key]  (transposed)
    __shared__ __align__(16) _Float16 Sbuf[4 * 16 * 320];

    int nh = blockIdx.x;
    int n = nh >> 3, h = nh & 7;
    int tid = threadIdx.x, lane = tid & 31, wave = tid >> 5;
    int hi = lane >> 4, ln15 = lane & 15;
    size_t base = (size_t)n * NSEQ * 192;
    const float scale = 0.35355339059327373f;          // 1/sqrt(8)

    // stage K rows and V (transposed) into LDS, zero-padded to 320 keys
    for (int s = tid; s < 320; s += 128) {
        uint32_t kq[4] = {0u, 0u, 0u, 0u};
        union { uint32_t u[4]; _Float16 hh[8]; } vv;
        vv.u[0] = vv.u[1] = vv.u[2] = vv.u[3] = 0u;
        if (s < NSEQ) {
            const _Float16* rp = qkv + base + (size_t)s * 192;
            const uint32_t* kp = (const uint32_t*)(rp + 64 + h * 8);
            const uint32_t* vp = (const uint32_t*)(rp + 128 + h * 8);
#pragma unroll
            for (int j = 0; j < 4; ++j) { kq[j] = kp[j]; vv.u[j] = vp[j]; }
        }
#pragma unroll
        for (int j = 0; j < 4; ++j)
            *(uint32_t*)(Kt + s * 8 + 2 * j) = kq[j];
#pragma unroll
        for (int d = 0; d < 8; ++d)
            Vt[d * 320 + s] = vv.hh[d];
    }
    __syncthreads();

    _Float16* Srow = Sbuf + wave * 16 * 320;

    for (int qi = 0; qi < 5; ++qi) {                   // 19 query tiles, 4 waves
        int qt = qi * 4 + wave;
        bool act = (qt < 19);                          // wave-uniform
        float rmax[8], invs[8];
        Frag qf;

        if (act) {
            // Q fragment: 16 queries x (head-dim 8, zero-padded to K=32)
#pragma unroll
            for (int j = 0; j < 8; ++j) qf.u[j] = 0u;
            int sq = qt * 16 + ln15;
            if (hi == 0 && sq < NSEQ) {
                const uint32_t* qp =
                    (const uint32_t*)(qkv + base + (size_t)sq * 192 + h * 8);
#pragma unroll
                for (int j = 0; j < 4; ++j) qf.u[j] = qp[j];
            }
#pragma unroll
            for (int r = 0; r < 8; ++r) rmax[r] = -3.0e38f;

            for (int jc = 0; jc < 19; ++jc) {          // score chunks of 16 keys
                Frag kf;
#pragma unroll
                for (int j = 0; j < 8; ++j) kf.u[j] = 0u;
                if (hi == 0) {
                    const uint32_t* kp = (const uint32_t*)(Kt + (jc * 16 + ln15) * 8);
#pragma unroll
                    for (int j = 0; j < 4; ++j) kf.u[j] = kp[j];
                }
                v8f z = {};
                v8f c = __builtin_amdgcn_wmma_f32_16x16x32_f16(
                    false, qf.v, false, kf.v, (short)0, z, false, false);
                int col = jc * 16 + ln15;
#pragma unroll
                for (int r = 0; r < 8; ++r) {
                    float v = c[r] * scale;
                    Srow[(r + 8 * hi) * 320 + col] = (_Float16)v;
                    if (col < NSEQ) rmax[r] = fmaxf(rmax[r], v);
                }
            }
#pragma unroll
            for (int m = 1; m <= 8; m <<= 1)
#pragma unroll
                for (int r = 0; r < 8; ++r)
                    rmax[r] = fmaxf(rmax[r], __shfl_xor(rmax[r], m, 32));
        }
        __syncthreads();

        if (act) {
            // softmax pass: exp + row-sum; pad cols [301,320) with zeros
#pragma unroll
            for (int r = 0; r < 8; ++r) {
                int rr = r + 8 * hi;
                float sum = 0.0f;
                for (int k2 = 0; k2 < 20; ++k2) {
                    int col = ln15 + (k2 << 4);
                    float ev = 0.0f;
                    if (col < NSEQ)
                        ev = __expf((float)Srow[rr * 320 + col] - rmax[r]);
                    Srow[rr * 320 + col] = (_Float16)ev;
                    sum += ev;
                }
#pragma unroll
                for (int m = 1; m <= 8; m <<= 1) sum += __shfl_xor(sum, m, 32);
                invs[r] = 1.0f / sum;
            }
        }
        __syncthreads();

        if (act) {
            // O = P @ V : ten 16x16x32 WMMAs over 320 padded keys
            v8f oacc = {};
            for (int jc2 = 0; jc2 < 10; ++jc2) {
                Frag pf, vf;
#pragma unroll
                for (int j = 0; j < 8; ++j) {
                    int kb = kb_of(j, hi) + jc2 * 32;
                    pf.u[j] = *(const uint32_t*)(Srow + ln15 * 320 + kb);
                    vf.u[j] = (ln15 < 8)
                                  ? *(const uint32_t*)(Vt + ln15 * 320 + kb)
                                  : 0u;
                }
                oacc = __builtin_amdgcn_wmma_f32_16x16x32_f16(
                    false, pf.v, false, vf.v, (short)0, oacc, false, false);
            }
            if (ln15 < 8) {
#pragma unroll
                for (int r = 0; r < 8; ++r) {
                    int s = qt * 16 + r + 8 * hi;
                    if (s < NSEQ)
                        o16[((size_t)n * NSEQ + s) * DM + h * 8 + ln15] =
                            (_Float16)(oacc[r] * invs[r]);
                }
            }
        }
        __syncthreads();
    }
}

// ---- cls pooling over edges + LN -------------------------------------------
__global__ __launch_bounds__(64) void pool_ln_kernel(
    const float* __restrict__ h32, const float* __restrict__ ng,
    const float* __restrict__ nb, float* __restrict__ featn) {
    int b = blockIdx.x, d = threadIdx.x;
    float acc = 0.0f;
    for (int e = 0; e < NE; ++e)
        acc += h32[(size_t)(b * NE + e) * NSEQ * DM + d];
    float f = acc * (1.0f / NE);
    __shared__ float sm[64];
    sm[d] = f;
    __syncthreads();
    float mu = 0.0f;
    for (int i = 0; i < 64; ++i) mu += sm[i];
    mu *= (1.0f / 64.0f);
    float var = 0.0f;
    for (int i = 0; i < 64; ++i) { float t = sm[i] - mu; var += t * t; }
    var *= (1.0f / 64.0f);
    featn[b * DM + d] = (f - mu) * rsqrtf(var + EPSLN) * ng[d] + nb[d];
}

// ---- tiny MLP head ----------------------------------------------------------
__global__ __launch_bounds__(64) void head_kernel(
    const float* __restrict__ featn, const float* __restrict__ w1,
    const float* __restrict__ b1, const float* __restrict__ w2,
    const float* __restrict__ b2, float* __restrict__ out) {
    int b = blockIdx.x, j = threadIdx.x;
    __shared__ float fin[64];
    __shared__ float g1[64];
    fin[j] = featn[b * DM + j];
    __syncthreads();
    float t = b1[j];
    for (int d = 0; d < 64; ++d) t += fin[d] * w1[j * 64 + d];
    g1[j] = geluf(t);
    __syncthreads();
    if (j < NCLS) {
        float o = b2[j];
        for (int d = 0; d < 64; ++d) o += g1[d] * w2[j * 64 + d];
        out[b * NCLS + j] = o;
    }
}

// ---- orchestration ----------------------------------------------------------
extern "C" void kernel_launch(void* const* d_in, const int* in_sizes, int n_in,
                              void* d_out, int out_size, void* d_ws, size_t ws_size,
                              hipStream_t stream) {
    const float* x       = (const float*)d_in[0];
    const float* embed_w = (const float*)d_in[1];
    const float* embed_b = (const float*)d_in[2];
    const float* eln_g   = (const float*)d_in[3];
    const float* eln_b   = (const float*)d_in[4];
    const float* cls     = (const float*)d_in[5];
    const float* qkv_w   = (const float*)d_in[6];
    const float* qkv_b   = (const float*)d_in[7];
    const float* out_w   = (const float*)d_in[8];
    const float* out_b   = (const float*)d_in[9];
    const float* ln1_g   = (const float*)d_in[10];
    const float* ln1_b   = (const float*)d_in[11];
    const float* ln2_g   = (const float*)d_in[12];
    const float* ln2_b   = (const float*)d_in[13];
    const float* ff1_w   = (const float*)d_in[14];
    const float* ff1_b   = (const float*)d_in[15];
    const float* ff2_w   = (const float*)d_in[16];
    const float* ff2_b   = (const float*)d_in[17];
    const float* norm_g  = (const float*)d_in[18];
    const float* norm_b  = (const float*)d_in[19];
    const float* h1_w    = (const float*)d_in[20];
    const float* h1_b    = (const float*)d_in[21];
    const float* h2_w    = (const float*)d_in[22];
    const float* h2_b    = (const float*)d_in[23];
    float* out = (float*)d_out;

    uint8_t* ws = (uint8_t*)d_ws;
    float*     h32     = (float*)(ws + 0);                       //  30,822,400 B
    _Float16*  y16     = (_Float16*)(ws + 30822400);             //  15,411,200 B
    _Float16*  qkv16   = (_Float16*)(ws + 46233600);             //  46,233,600 B
    _Float16*  o16     = (_Float16*)(ws + 92467200);             //  15,411,200 B
    _Float16*  ff116   = (_Float16*)(ws + 107878400);            //  61,644,800 B
    _Float16*  qkvw16  = (_Float16*)(ws + 169523200);            //      98,304 B
    _Float16*  outw16  = (_Float16*)(ws + 169621504);            //      32,768 B
    _Float16*  ff1w16  = (_Float16*)(ws + 169654272);            //     131,072 B
    _Float16*  ff2w16  = (_Float16*)(ws + 169785344);            //     131,072 B
    float*     featn   = (float*)(ws + 169916416);               //       4,096 B

    const int MB = MTOT / 8;               // 15050 row-blocks (8 waves each)
    const int GT = (MTOT + 127) / 128;     // 941 GEMM M-tiles (128 rows each)

    cvt_f16_kernel<<<(NLAY*3*DM*DM + 255) / 256, 256, 0, stream>>>(qkv_w, qkvw16, NLAY*3*DM*DM);
    cvt_f16_kernel<<<(NLAY*DM*DM   + 255) / 256, 256, 0, stream>>>(out_w, outw16, NLAY*DM*DM);
    cvt_f16_kernel<<<(NLAY*FFD*DM  + 255) / 256, 256, 0, stream>>>(ff1_w, ff1w16, NLAY*FFD*DM);
    cvt_f16_kernel<<<(NLAY*DM*FFD  + 255) / 256, 256, 0, stream>>>(ff2_w, ff2w16, NLAY*DM*FFD);

    embed_ln_pos_kernel<<<MB, 256, 0, stream>>>(x, embed_w, embed_b, eln_g, eln_b, cls, h32);

    for (int l = 0; l < NLAY; ++l) {
        ln_kernel<<<MB, 256, 0, stream>>>(h32, ln1_g + l * DM, ln1_b + l * DM, y16);
        gemm_wmma_kernel<64, 0><<<dim3(GT, 3), 128, 0, stream>>>(
            y16, qkvw16 + l * 3 * DM * DM, qkv_b + l * 3 * DM,
            nullptr, qkv16, nullptr, MTOT, 3 * DM);
        attn_kernel<<<NROW * NH, 128, 0, stream>>>(qkv16, o16);
        gemm_wmma_kernel<64, 1><<<dim3(GT, 1), 128, 0, stream>>>(
            o16, outw16 + l * DM * DM, out_b + l * DM,
            h32, nullptr, h32, MTOT, DM);
        ln_kernel<<<MB, 256, 0, stream>>>(h32, ln2_g + l * DM, ln2_b + l * DM, y16);
        gemm_wmma_kernel<64, 2><<<dim3(GT, 4), 128, 0, stream>>>(
            y16, ff1w16 + l * FFD * DM, ff1_b + l * FFD,
            nullptr, ff116, nullptr, MTOT, FFD);
        gemm_wmma_kernel<256, 1><<<dim3(GT, 1), 128, 0, stream>>>(
            ff116, ff2w16 + l * DM * FFD, ff2_b + l * DM,
            h32, nullptr, h32, MTOT, DM);
    }

    pool_ln_kernel<<<NB, 64, 0, stream>>>(h32, norm_g, norm_b, featn);
    head_kernel<<<NB, 64, 0, stream>>>(featn, h1_w, h1_b, h2_w, h2_b, out);
}
